// XLSTMMemoryCell_11785390260729
// MI455X (gfx1250) — matrix operand order repaired and checked by reading, not compiled, not run on hardware
//
#include <hip/hip_runtime.h>
#include <hip/hip_bf16.h>

// Problem sizes (fixed by the reference).
#define Bsz   64
#define Tn    512
#define Dn    256
#define Un    512
#define Mn    64
#define KTOT  832      // D + U + M = 256 + 512 + 64
#define NKT   26       // KTOT / 32 k-tiles
#define NCT   128      // 4U / 16 column tiles (gate-interleaved)

typedef __attribute__((ext_vector_type(8)))  float  v8f;
typedef __attribute__((ext_vector_type(16))) __bf16 v16bf;

// ---------------------------------------------------------------------------
// WMMA fragment index maps (CDNA5 ISA §7.12.2, wave32)
// A (16x32 bf16): lane L holds row m=L&15; element e (0..15):
//   lanes 0-15 : e0..7 -> K 0..7,  e8..15 -> K 16..23
//   lanes16-31 : e0..7 -> K 8..15, e8..15 -> K 24..31
// B (32x16 bf16): lane L holds col n=L&15; element e -> same K map as A
// C/D (16x16 f32): vgpr r, lane L -> (m = r + (L>=16)*8, n = L&15)
// ---------------------------------------------------------------------------
__device__ __forceinline__ int a_kk(int lane, int e) {
  int hi = (lane >> 4) & 1;
  return (e & 7) + hi * 8 + ((e >> 3) << 4);
}
__device__ __forceinline__ int b_kk(int lane, int e) {
  return (e & 7) + ((lane >> 4) & 1) * 8 + ((e >> 3) << 4);
}
// Inverse of a_kk: given kk in [0,32) -> (lane-half, element)
__device__ __forceinline__ void inv_a(int kk, int& lanehi, int& e) {
  lanehi = (kk >> 3) & 1;
  e = (kk & 7) + ((kk >> 4) << 3);
}

// Opaque zero offset: defeats LICM (loads can't be hoisted out of the time
// loop -> no scratch spills) while PRESERVING pointer provenance, so the
// compiler still infers addrspace(1) and emits global_load (not flat_load).
__device__ __forceinline__ size_t opaque_zero() {
  unsigned long long z = 0;
  asm volatile("" : "+s"(z));
  return (size_t)z;
}

// ---------------------------------------------------------------------------
// Prep kernel 1: combined weights  Wbig[KTOT][4U]  ->  B-fragments, bf16.
// Column interleave: global col j = ub*64 + g*16 + n  (u = ub*16+n, gate g).
// Rows: 0..255 = Wx, 256..767 = Uh, 768..831 = W_mem (i-gate cols only).
// Layout: WbigB[ctile(128)][kt(26)][lane*16+e], 1 frag = 512 bf16 = 1 KiB.
// ---------------------------------------------------------------------------
__global__ void build_wbig_kernel(const float* __restrict__ Wi, const float* __restrict__ Wf,
                                  const float* __restrict__ Wc, const float* __restrict__ Wo,
                                  const float* __restrict__ Ui, const float* __restrict__ Uf,
                                  const float* __restrict__ Uc, const float* __restrict__ Uo,
                                  const float* __restrict__ Wm, __bf16* __restrict__ out) {
  int wid  = (int)((blockIdx.x * blockDim.x + threadIdx.x) >> 5);
  int lane = (int)(threadIdx.x & 31);
  if (wid >= NCT * NKT) return;
  int ctile = wid / NKT, kt = wid % NKT;
  int ub = ctile >> 2, g = ctile & 3;
  int u = ub * 16 + (lane & 15);
  const float* Wg[4] = {Wi, Wf, Wc, Wo};
  const float* Ug[4] = {Ui, Uf, Uc, Uo};
  v16bf vals;
#pragma unroll
  for (int e = 0; e < 16; ++e) {
    int k = kt * 32 + b_kk(lane, e);
    float v;
    if (k < Dn)            v = Wg[g][(size_t)k * Un + u];
    else if (k < Dn + Un)  v = Ug[g][(size_t)(k - Dn) * Un + u];
    else                   v = (g == 0) ? Wm[(size_t)(k - Dn - Un) * Un + u] : 0.f;
    vals[e] = (__bf16)v;
  }
  *(v16bf*)(out + (size_t)wid * 512 + lane * 16) = vals;
}

// ---------------------------------------------------------------------------
// Prep kernel 2: x (f32 [B,T,D]) -> bf16 A-fragments.
// Layout: xA[slice(4)][t(512)][kt(8)][lane*16+e]; 16384 fragments.
// ---------------------------------------------------------------------------
__global__ void build_xa_kernel(const float* __restrict__ x, __bf16* __restrict__ out) {
  int wid  = (int)((blockIdx.x * blockDim.x + threadIdx.x) >> 5);
  int lane = (int)(threadIdx.x & 31);
  if (wid >= 4 * Tn * 8) return;
  int slice = wid / (Tn * 8);
  int rem   = wid % (Tn * 8);
  int t = rem / 8, kt = rem % 8;
  int m = lane & 15;
  int b = slice * 16 + m;
  v16bf vals;
#pragma unroll
  for (int e = 0; e < 16; ++e) {
    int k = kt * 32 + a_kk(lane, e);   // < 256 for kt < 8
    vals[e] = (__bf16)x[((size_t)b * Tn + t) * Dn + k];
  }
  *(v16bf*)(out + (size_t)wid * 512 + lane * 16) = vals;
}

// ---------------------------------------------------------------------------
// Prep kernel 3: U_mem (f32 [U,M]) -> bf16 B-fragments.
// Layout: UmemB[ctm(4)][kt(16)][lane*16+e]; 64 fragments.
// ---------------------------------------------------------------------------
__global__ void build_umem_kernel(const float* __restrict__ Um, __bf16* __restrict__ out) {
  int wid  = (int)((blockIdx.x * blockDim.x + threadIdx.x) >> 5);
  int lane = (int)(threadIdx.x & 31);
  if (wid >= 4 * 16) return;
  int ctm = wid >> 4, kt = wid & 15;
  int mm = ctm * 16 + (lane & 15);
  v16bf vals;
#pragma unroll
  for (int e = 0; e < 16; ++e) {
    int k = kt * 32 + b_kk(lane, e);   // < 512
    vals[e] = (__bf16)Um[(size_t)k * Mn + mm];
  }
  *(v16bf*)(out + (size_t)wid * 512 + lane * 16) = vals;
}

// 8 gate-tile WMMAs for one k-tile: load 4 B-fragments (s_clause), then 4
// WMMAs, twice.  A-fragment is reused across all 8 tiles.
__device__ __forceinline__ void gate_mma8(const v16bf afrag, const __bf16* __restrict__ wb,
                                          int ub0, int kt, int lane, v8f acc[2][4]) {
#pragma unroll
  for (int q = 0; q < 2; ++q) {
    v16bf bfr[4];
#pragma unroll
    for (int g = 0; g < 4; ++g)
      bfr[g] = *(const v16bf*)(wb + ((size_t)((ub0 + q) * 4 + g) * NKT + kt) * 512 + lane * 16);
#pragma unroll
    for (int g = 0; g < 4; ++g)
      acc[q][g] = __builtin_amdgcn_wmma_f32_16x16x32_bf16(
          false, afrag, false, bfr[g], (short)0, acc[q][g], false, false);
  }
}

// ---------------------------------------------------------------------------
// Scan kernel: 4 blocks (one per 16-row batch slice), 512 threads = 16 waves.
// Wave w owns u-blocks {2w, 2w+1}; per step, per u-block it computes all 4
// gate tiles (i,f,c,o) of z = [x_t|h|mem] @ Wbig with 26 k-tiles of WMMA
// bf16 16x16x32, does the gate math in registers (c-state never leaves
// VGPRs), writes h to the output and scatters bf16 h into the
// double-buffered LDS A-fragment store for the next step.  Waves 0..3 then
// apply mem += 0.1*(h_new @ U_mem) via WMMA.
// ---------------------------------------------------------------------------
__global__ __launch_bounds__(512, 1)
void xlstm_scan_kernel(const __bf16* __restrict__ WbigB,
                       const __bf16* __restrict__ xA,
                       const __bf16* __restrict__ UmemB,
                       const float* __restrict__ bi_p, const float* __restrict__ bf_p,
                       const float* __restrict__ bc_p, const float* __restrict__ bo_p,
                       float* __restrict__ out) {
  __shared__ __align__(16) __bf16 swzh[2][16][512];   // h state, A-frag layout, dbl-buffered
  __shared__ __align__(16) __bf16 swzm[2][2][512];    // mem state (bf16), A-frag layout
  __shared__ __align__(16) float  memf[16 * Mn];      // mem state f32 [16 rows][64]

  const int slice = (int)blockIdx.x;
  const int tid   = (int)threadIdx.x;
  const int lane  = tid & 31;
  const int w     = tid >> 5;     // 0..15
  const int n     = lane & 15;
  const int hi    = lane >> 4;

  // h = c = mem = 0 at t = 0
  for (int i = tid; i < 2 * 16 * 512; i += 512) ((__bf16*)swzh)[i] = (__bf16)0.f;
  for (int i = tid; i < 2 * 2 * 512;  i += 512) ((__bf16*)swzm)[i] = (__bf16)0.f;
  for (int i = tid; i < 16 * Mn;      i += 512) memf[i] = 0.f;
  __syncthreads();

  const int ub0 = w * 2;
  float bia[2], bfa[2], bca[2], boa[2];
#pragma unroll
  for (int q = 0; q < 2; ++q) {
    int u = (ub0 + q) * 16 + n;
    bia[q] = bi_p[u]; bfa[q] = bf_p[u]; bca[q] = bc_p[u]; boa[q] = bo_p[u];
  }

  v8f cstate[2] = {};   // cell state lives in VGPRs for the whole scan
  const __bf16* xa_slice = xA + (size_t)slice * Tn * 8 * 512;

  for (int t = 0; t < Tn; ++t) {
    const int cur = t & 1, nxt = cur ^ 1;
    v8f acc[2][4] = {};

    // Fresh (un-hoistable, but still addrspace(1)) bases every timestep.
    const __bf16* wb   = WbigB + opaque_zero();
    const __bf16* xa_t = xa_slice + (size_t)t * 8 * 512 + opaque_zero();

    if (t + 1 < Tn) {   // warm L2/WGP$ for next step's x fragments
      const __bf16* pf = xa_slice + ((size_t)(t + 1) * 8 + (w & 7)) * 512 + lane * 16;
      __builtin_prefetch(pf, 0, 3);
    }

    // ---- phase A: z = [x_t | h | mem] @ Wbig, K = 832 ----
    // Region 1: x contribution (A-frags from global), kt 0..7.
#pragma unroll 2
    for (int kt = 0; kt < 8; ++kt) {
      v16bf afrag = *(const v16bf*)(xa_t + (size_t)kt * 512 + lane * 16);
      gate_mma8(afrag, wb, ub0, kt, lane, acc);
    }
    // Region 2: h contribution (A-frags from LDS), kt 8..23.
#pragma unroll 2
    for (int kt = 8; kt < 24; ++kt) {
      v16bf afrag = *(const v16bf*)(&swzh[cur][kt - 8][lane * 16]);
      gate_mma8(afrag, wb, ub0, kt, lane, acc);
    }
    // Region 3: mem contribution (A-frags from LDS), kt 24..25.
#pragma unroll
    for (int kt = 24; kt < 26; ++kt) {
      v16bf afrag = *(const v16bf*)(&swzm[cur][kt - 24][lane * 16]);
      gate_mma8(afrag, wb, ub0, kt, lane, acc);
    }

    // ---- gates + state update (registers), write h, scatter h -> LDS ----
#pragma unroll
    for (int q = 0; q < 2; ++q) {
      const int ub = ub0 + q;
#pragma unroll
      for (int r = 0; r < 8; ++r) {
        const int m = r + hi * 8;
        float zi = acc[q][0][r] + bia[q];
        float zf = acc[q][1][r] + bfa[q];
        float zc = acc[q][2][r] + bca[q];
        float zo = acc[q][3][r] + boa[q];
        float ig = 1.f / (1.f + __expf(-zi));   // mem_read already folded into zi
        float fg = 1.f / (1.f + __expf(-zf));
        float cc = tanhf(zc);
        float og = 1.f / (1.f + __expf(-zo));
        float cn = fg * cstate[q][r] + ig * cc;
        cstate[q][r] = cn;
        float hn = og * tanhf(cn);

        const int u = ub * 16 + n;
        const int b = slice * 16 + m;
        out[((size_t)b * Tn + t) * Un + u] = hn;

        int lh, ee; inv_a(u & 31, lh, ee);
        swzh[nxt][u >> 5][(m + lh * 16) * 16 + ee] = (__bf16)hn;
      }
    }
    __syncthreads();

    // ---- phase B: mem += 0.1 * (h_new @ U_mem), waves 0..3 ----
    if (w < 4) {
      const __bf16* um = UmemB + opaque_zero();
      v8f macc = {};
#pragma unroll
      for (int kt4 = 0; kt4 < 4; ++kt4) {
        v16bf afr[4], bfr[4];
#pragma unroll
        for (int j = 0; j < 4; ++j) {
          const int kt = kt4 * 4 + j;
          afr[j] = *(const v16bf*)(&swzh[nxt][kt][lane * 16]);
          bfr[j] = *(const v16bf*)(um + ((size_t)w * 16 + kt) * 512 + lane * 16);
        }
#pragma unroll
        for (int j = 0; j < 4; ++j)
          macc = __builtin_amdgcn_wmma_f32_16x16x32_bf16(
              false, afr[j], false, bfr[j], (short)0, macc, false, false);
      }
#pragma unroll
      for (int r = 0; r < 8; ++r) {
        const int m  = r + hi * 8;
        const int mm = w * 16 + n;
        float mv = memf[m * Mn + mm] + 0.1f * macc[r];
        memf[m * Mn + mm] = mv;
        int lh, ee; inv_a(mm & 31, lh, ee);
        swzm[nxt][mm >> 5][(m + lh * 16) * 16 + ee] = (__bf16)mv;
      }
    }
    __syncthreads();
  }
}

// ---------------------------------------------------------------------------
extern "C" void kernel_launch(void* const* d_in, const int* in_sizes, int n_in,
                              void* d_out, int out_size, void* d_ws, size_t ws_size,
                              hipStream_t stream) {
  (void)in_sizes; (void)n_in; (void)out_size; (void)ws_size;
  const float* x   = (const float*)d_in[0];
  const float* Wi  = (const float*)d_in[1];
  const float* Wf  = (const float*)d_in[2];
  const float* Wc  = (const float*)d_in[3];
  const float* Wo  = (const float*)d_in[4];
  const float* Ui  = (const float*)d_in[5];
  const float* Uf  = (const float*)d_in[6];
  const float* Uc  = (const float*)d_in[7];
  const float* Uo  = (const float*)d_in[8];
  const float* Wm  = (const float*)d_in[9];
  const float* Um  = (const float*)d_in[10];
  const float* bi  = (const float*)d_in[11];
  const float* bf  = (const float*)d_in[12];
  const float* bc  = (const float*)d_in[13];
  const float* bo  = (const float*)d_in[14];

  char* ws = (char*)d_ws;
  const size_t wbig_bytes = (size_t)NCT * NKT * 512 * sizeof(__bf16);   // 3,407,872
  const size_t xa_bytes   = (size_t)4 * Tn * 8 * 512 * sizeof(__bf16);  // 16,777,216
  __bf16* WbigB = (__bf16*)ws;
  __bf16* xAB   = (__bf16*)(ws + wbig_bytes);
  __bf16* UmemB = (__bf16*)(ws + wbig_bytes + xa_bytes);                // 65,536 B

  // Pre-swizzle weights and inputs into WMMA fragment layouts (bf16).
  build_wbig_kernel<<<(NCT * NKT + 7) / 8, 256, 0, stream>>>(Wi, Wf, Wc, Wo,
                                                             Ui, Uf, Uc, Uo, Wm, WbigB);
  build_xa_kernel<<<(4 * Tn * 8) / 8, 256, 0, stream>>>(x, xAB);
  build_umem_kernel<<<8, 256, 0, stream>>>(Um, UmemB);

  // Sequential scan: one workgroup per 16-row batch slice (WMMA M-tile).
  xlstm_scan_kernel<<<4, 512, 0, stream>>>(WbigB, xAB, UmemB, bi, bf, bc, bo,
                                           (float*)d_out);
}
